// Graph_Convolution_23106924052606
// MI455X (gfx1250) — matrix operand order repaired and checked by reading, not compiled, not run on hardware
//
#include <hip/hip_runtime.h>
#include <hip/hip_bf16.h>

// ---------------------------------------------------------------------------
// Sizes (fixed by the reference)
// ---------------------------------------------------------------------------
#define BATCH 1024
#define NNODE 39            // 39x39 matrices, 39 nodes per graph
#define PIX   (NNODE*NNODE) // 1521
#define EPER  256
#define E1    (BATCH*EPER)      // 262144 graph edges
#define NTOT  (BATCH*NNODE)     // 39936 nodes == GEMM M (divisible by 16)
#define ETOT  (E1 + NTOT)       // + self loops = 302080
#define F1    256               // GAT1 width (8 heads x 32)
#define H1N   8
#define D1    32
#define F2    64                // GAT2 width (1 head x 64)
#define K1P   96                // 78 padded to 3x32
#define K2P   256

typedef __attribute__((ext_vector_type(16))) _Float16 v16h;
typedef __attribute__((ext_vector_type(8)))  _Float16 v8h;
typedef __attribute__((ext_vector_type(8)))  float    v8f;

// ---------------------------------------------------------------------------
// Order-preserving float<->uint encoding for atomicMax-based segment max
// ---------------------------------------------------------------------------
__device__ __forceinline__ unsigned fenc(float f) {
    unsigned u = __float_as_uint(f);
    return (u & 0x80000000u) ? ~u : (u | 0x80000000u);
}
__device__ __forceinline__ float fdec(unsigned u) {
    return (u & 0x80000000u) ? __uint_as_float(u & 0x7FFFFFFFu) : __uint_as_float(~u);
}

// ---------------------------------------------------------------------------
// Kernel 1: fused NaN-clean + conv1(1x1)+BN1+ReLU + conv2(3x3,pad1)+BN2 + GRU.
// One block per batch image. conv1 plane kept in LDS; conv2 produced one row
// at a time (GRU consumes rows in order). Writes masked GRU hidden states
// directly into the f16 GAT feature matrix, columns [39,78).
// ---------------------------------------------------------------------------
__global__ __launch_bounds__(256) void conv_gru_kernel(
    const float* __restrict__ ve, const float* __restrict__ ac,
    const float* __restrict__ man, const float* __restrict__ mask,
    const float* __restrict__ c1w, const float* __restrict__ c1b,
    const float* __restrict__ bn1g, const float* __restrict__ bn1b,
    const float* __restrict__ bn1m, const float* __restrict__ bn1v,
    const float* __restrict__ c2w, const float* __restrict__ c2b,
    const float* __restrict__ bn2g, const float* __restrict__ bn2b,
    const float* __restrict__ bn2m, const float* __restrict__ bn2v,
    const float* __restrict__ wih, const float* __restrict__ whh,
    const float* __restrict__ bih, const float* __restrict__ bhh,
    _Float16* __restrict__ gFeat)
{
    __shared__ float sc1[8][PIX];      // conv1 output plane   (~49 KB)
    __shared__ float sc2[16][NNODE];   // one conv2 output row
    __shared__ float sw2[16 * 8 * 9];  // conv2 weights
    __shared__ float sw1[24], sb1[8], s1s[8], s1o[8];
    __shared__ float sb2[16], s2s[16], s2o[16];
    __shared__ float swih[48];

    const int b   = blockIdx.x;
    const int tid = threadIdx.x;

    for (int i = tid; i < 1152; i += 256) sw2[i] = c2w[i];
    if (tid < 24) sw1[tid] = c1w[tid];
    if (tid < 48) swih[tid] = wih[tid];
    if (tid < 8) {
        sb1[tid] = c1b[tid];
        float s = bn1g[tid] * rsqrtf(bn1v[tid] + 1e-5f);
        s1s[tid] = s; s1o[tid] = bn1b[tid] - bn1m[tid] * s;
    }
    if (tid < 16) {
        sb2[tid] = c2b[tid];
        float s = bn2g[tid] * rsqrtf(bn2v[tid] + 1e-5f);
        s2s[tid] = s; s2o[tid] = bn2b[tid] - bn2m[tid] * s;
    }
    __syncthreads();

    const float* mb = man + (size_t)b * PIX;
    const float* ab = ac  + (size_t)b * PIX;
    const float* vb = ve  + (size_t)b * PIX;

    // conv1 (channel order: stack([man, ac, ve]))
    for (int i = tid; i < 8 * PIX; i += 256) {
        int co = i / PIX, p = i % PIX;
        float x0 = mb[p]; if (x0 != x0) x0 = 0.f;
        float x1 = ab[p]; if (x1 != x1) x1 = 0.f;
        float x2 = vb[p]; if (x2 != x2) x2 = 0.f;
        float s = sb1[co] + sw1[co*3]*x0 + sw1[co*3+1]*x1 + sw1[co*3+2]*x2;
        float t = s1s[co] * s + s1o[co];
        sc1[co][p] = fmaxf(t, 0.f);
    }
    __syncthreads();

    const int w   = tid;  // GRU column (threads 0..38 active in GRU phase)
    float hprev   = 0.f;
    float msk     = (w < NNODE) ? mask[b * NNODE + w] : 0.f;
    float rb0 = bih[0], rb1 = bih[1], rb2 = bih[2];
    float hb0 = bhh[0], hb1 = bhh[1], hb2 = bhh[2];
    float wh0 = whh[0], wh1 = whh[1], wh2 = whh[2];

    for (int hr = 0; hr < NNODE; ++hr) {
        // conv2 + BN2 for row hr
        for (int i = tid; i < 16 * NNODE; i += 256) {
            int co = i / NNODE, x = i % NNODE;
            float s = sb2[co];
            const float* wp = &sw2[co * 72];
            #pragma unroll
            for (int ci = 0; ci < 8; ++ci) {
                const float* cp = sc1[ci];
                #pragma unroll
                for (int dy = -1; dy <= 1; ++dy) {
                    int yy = hr + dy; if ((unsigned)yy >= NNODE) continue;
                    const float* rowp = cp + yy * NNODE;
                    #pragma unroll
                    for (int dx = -1; dx <= 1; ++dx) {
                        int xx = x + dx; if ((unsigned)xx >= NNODE) continue;
                        s += wp[ci*9 + (dy+1)*3 + (dx+1)] * rowp[xx];
                    }
                }
            }
            sc2[co][x] = s2s[co] * s + s2o[co];
        }
        __syncthreads();
        if (w < NNODE) {
            float g0 = rb0, g1 = rb1, g2 = rb2;
            #pragma unroll
            for (int i = 0; i < 16; ++i) {
                float xv = sc2[i][w];
                g0 += swih[i] * xv; g1 += swih[16+i] * xv; g2 += swih[32+i] * xv;
            }
            float r = 1.f / (1.f + __expf(-(g0 + hb0 + wh0 * hprev)));
            float z = 1.f / (1.f + __expf(-(g1 + hb1 + wh1 * hprev)));
            float n = tanhf(g2 + r * (hb2 + wh2 * hprev));
            hprev = (1.f - z) * n + z * hprev;
            gFeat[(size_t)(b * NNODE + w) * K1P + 39 + hr] = (_Float16)(hprev * msk);
        }
        __syncthreads();
    }
}

// ---------------------------------------------------------------------------
// Kernel 2: masked man -> f16 feature columns [0,39); zero pad [78,96).
// ---------------------------------------------------------------------------
__global__ void man_feat_kernel(const float* __restrict__ man,
                                const float* __restrict__ mask,
                                _Float16* __restrict__ gFeat)
{
    size_t t = (size_t)blockIdx.x * 256 + threadIdx.x;
    if (t >= (size_t)NTOT * K1P) return;
    int n = (int)(t / K1P), k = (int)(t % K1P);
    if (k >= 39 && k < 78) return;           // written by conv_gru_kernel
    float v = 0.f;
    if (k < 39) {
        int b = n / NNODE, j = n % NNODE;
        float mv = man[(size_t)b * PIX + k * NNODE + j];
        if (mv != mv) mv = 0.f;
        v = mv * mask[n];
    }
    gFeat[t] = (_Float16)v;
}

// ---------------------------------------------------------------------------
// Kernel 3: weight f32 [K0 x N] row-major -> f16 transposed [N x Kpad]
// ---------------------------------------------------------------------------
__global__ void convert_wT_kernel(const float* __restrict__ w,
                                  _Float16* __restrict__ wT,
                                  int K0, int N, int Kpad)
{
    int t = blockIdx.x * 256 + threadIdx.x;
    if (t >= N * Kpad) return;
    int n = t / Kpad, k = t % Kpad;
    wT[t] = (_Float16)(k < K0 ? w[(size_t)k * N + n] : 0.f);
}

// ---------------------------------------------------------------------------
// Kernel 4: dual WMMA GEMM, 16x64 output strip per wave (4 N-tiles for BOTH
// projections, 8 accumulators). The shared A fragment is loaded once per
// K-step and feeds 8 v_wmma_f32_16x16x32_f16 ops -> 4x less A traffic than
// one-tile-per-wave. A: [M x Kpad] f16 row-major. BT: [N x Kpad] f16.
// K order: contiguous per lane half (k = 16*hi + s) for both A and B, not the ISA table order.
// ---------------------------------------------------------------------------
__global__ __launch_bounds__(32) void wmma_dual_gemm_kernel(
    const _Float16* __restrict__ A,
    const _Float16* __restrict__ BlT, const _Float16* __restrict__ BrT,
    const float* __restrict__ biasL, const float* __restrict__ biasR,
    float* __restrict__ Cl, float* __restrict__ Cr,
    int N, int Kpad)
{
    const int tm   = blockIdx.x;
    const int tn   = blockIdx.y;   // 64-column strip
    const int lane = threadIdx.x;
    const int hi   = lane >> 4;    // 0 = lanes 0..15, 1 = lanes 16..31
    const int l15  = lane & 15;

    const _Float16* arow = A   + (size_t)(tm * 16 + l15) * Kpad;
    const _Float16* bl0  = BlT + (size_t)(tn * 64 + l15) * Kpad;
    const _Float16* br0  = BrT + (size_t)(tn * 64 + l15) * Kpad;
    const size_t    bstp = (size_t)16 * Kpad;   // 16 columns of BT

    v8f accL[4] = {v8f{}, v8f{}, v8f{}, v8f{}};
    v8f accR[4] = {v8f{}, v8f{}, v8f{}, v8f{}};

    for (int kb = 0; kb < Kpad; kb += 32) {
        // A fragment: lanes<16 hold K = kb+[0..15]; lanes>=16 hold kb+[16..31]
        union { v16h v; v8h h[2]; } ua;
        const int aoff = kb + hi * 16;
        ua.h[0] = *(const v8h*)(arow + aoff);
        ua.h[1] = *(const v8h*)(arow + aoff + 8);
        // B fragments: lanes<16 hold K = kb+[0..15]; lanes>=16 hold kb+[16..31]
        const int boff = kb + hi * 16;
        #pragma unroll
        for (int c = 0; c < 4; ++c) {
            v16h bl = *(const v16h*)(bl0 + c * bstp + boff);
            v16h br = *(const v16h*)(br0 + c * bstp + boff);
            accL[c] = __builtin_amdgcn_wmma_f32_16x16x32_f16(
                false, ua.v, false, bl, (short)0, accL[c], false, false);
            accR[c] = __builtin_amdgcn_wmma_f32_16x16x32_f16(
                false, ua.v, false, br, (short)0, accR[c], false, false);
        }
    }
    // C/D layout: col = lane&15; VGPR r -> row r (+8 for lanes>=16)
    const int mb = tm * 16 + hi * 8;
    #pragma unroll
    for (int c = 0; c < 4; ++c) {
        const int n  = tn * 64 + c * 16 + l15;
        const float bL = biasL[n], bR = biasR[n];
        #pragma unroll
        for (int r = 0; r < 8; ++r) {
            size_t idx = (size_t)(mb + r) * N + n;
            Cl[idx] = accL[c][r] + bL;
            Cr[idx] = accR[c][r] + bR;
        }
    }
}

// ---------------------------------------------------------------------------
// Utility fills
// ---------------------------------------------------------------------------
__global__ void fill_u32_kernel(unsigned* __restrict__ p, unsigned v, size_t n)
{
    size_t t = (size_t)blockIdx.x * 256 + threadIdx.x;
    if (t < n) p[t] = v;
}
__global__ void init_bias_kernel(float* __restrict__ out,
                                 const float* __restrict__ bias,
                                 size_t rows, int F)
{
    size_t t = (size_t)blockIdx.x * 256 + threadIdx.x;
    if (t < rows * (size_t)F) out[t] = bias[t % F];
}

// ---------------------------------------------------------------------------
// GATv2 edge passes (generic over H heads, D dims per head)
// ---------------------------------------------------------------------------
__device__ __forceinline__ void edge_ends(const int* __restrict__ eidx,
                                          int e, int& src, int& dst)
{
    if (e < E1) { src = eidx[e]; dst = eidx[E1 + e]; }
    else        { src = e - E1; dst = e - E1; }       // self loops
}

__global__ void gat_logit_max_kernel(const int* __restrict__ eidx,
                                     const float* __restrict__ xl,
                                     const float* __restrict__ xr,
                                     const float* __restrict__ att,
                                     float* __restrict__ logit,
                                     unsigned* __restrict__ mmax,
                                     int H, int D)
{
    size_t t = (size_t)blockIdx.x * 256 + threadIdx.x;
    if (t >= (size_t)ETOT * H) return;
    int e = (int)(t / H), h = (int)(t % H);
    int src, dst; edge_ends(eidx, e, src, dst);
    const int F = H * D;
    const float* pl = xl + (size_t)src * F + h * D;
    const float* pr = xr + (size_t)dst * F + h * D;
    const float* pa = att + h * D;
    float s = 0.f;
    for (int d = 0; d < D; ++d) {
        float v = pl[d] + pr[d];
        v = v > 0.f ? v : 0.2f * v;          // leaky_relu(0.2)
        s += pa[d] * v;
    }
    logit[t] = s;
    atomicMax(&mmax[(size_t)dst * H + h], fenc(s));
}

__global__ void gat_expsum_kernel(const int* __restrict__ eidx,
                                  float* __restrict__ logit,   // in: logit, out: a
                                  const unsigned* __restrict__ mmax,
                                  float* __restrict__ ssum,
                                  int H)
{
    size_t t = (size_t)blockIdx.x * 256 + threadIdx.x;
    if (t >= (size_t)ETOT * H) return;
    int e = (int)(t / H), h = (int)(t % H);
    int src, dst; edge_ends(eidx, e, src, dst);
    float a = __expf(logit[t] - fdec(mmax[(size_t)dst * H + h]));
    logit[t] = a;
    atomicAdd(&ssum[(size_t)dst * H + h], a);
}

// float4-vectorized scatter: 4 payload elements per thread
__global__ void gat_aggregate_kernel(const int* __restrict__ eidx,
                                     const float* __restrict__ a,
                                     const float* __restrict__ ssum,
                                     const float* __restrict__ xl,
                                     float* __restrict__ out,
                                     int H, int D)
{
    const int F = H * D, Q = F >> 2;
    size_t t = (size_t)blockIdx.x * 256 + threadIdx.x;
    if (t >= (size_t)ETOT * Q) return;
    int e = (int)(t / Q), f = ((int)(t % Q)) << 2;
    int h = f / D;                               // D % 4 == 0 -> h uniform in group
    int src, dst; edge_ends(eidx, e, src, dst);
    float alpha = a[(size_t)e * H + h] / (ssum[(size_t)dst * H + h] + 1e-16f);
    const float4 v = *(const float4*)(xl + (size_t)src * F + f);
    float* po = out + (size_t)dst * F + f;
    atomicAdd(po + 0, v.x * alpha);
    atomicAdd(po + 1, v.y * alpha);
    atomicAdd(po + 2, v.z * alpha);
    atomicAdd(po + 3, v.w * alpha);
}

// ---------------------------------------------------------------------------
// ELU (bias already folded into accumulator init) -> f16 for next GEMM
// ---------------------------------------------------------------------------
__global__ void elu_f16_kernel(const float* __restrict__ in,
                               _Float16* __restrict__ out, size_t n)
{
    size_t t = (size_t)blockIdx.x * 256 + threadIdx.x;
    if (t >= n) return;
    float v = in[t];
    out[t] = (_Float16)(v > 0.f ? v : (__expf(v) - 1.f));
}

// ---------------------------------------------------------------------------
// Host orchestration
// ---------------------------------------------------------------------------
static inline size_t cdivz(size_t a, size_t b) { return (a + b - 1) / b; }

extern "C" void kernel_launch(void* const* d_in, const int* in_sizes, int n_in,
                              void* d_out, int out_size, void* d_ws, size_t ws_size,
                              hipStream_t stream)
{
    (void)in_sizes; (void)n_in; (void)out_size; (void)ws_size;

    const int*   eidx  = (const int*)  d_in[0];
    const float* ve    = (const float*)d_in[1];
    const float* ac    = (const float*)d_in[2];
    const float* man   = (const float*)d_in[3];
    const float* mask  = (const float*)d_in[4];
    //            d_in[5] = graph_matrix (dead input)
    const float* c1w   = (const float*)d_in[6];
    const float* c1b   = (const float*)d_in[7];
    const float* bn1g  = (const float*)d_in[8];
    const float* bn1b  = (const float*)d_in[9];
    const float* bn1m  = (const float*)d_in[10];
    const float* bn1v  = (const float*)d_in[11];
    const float* c2w   = (const float*)d_in[12];
    const float* c2b   = (const float*)d_in[13];
    const float* bn2g  = (const float*)d_in[14];
    const float* bn2b  = (const float*)d_in[15];
    const float* bn2m  = (const float*)d_in[16];
    const float* bn2v  = (const float*)d_in[17];
    const float* gwih  = (const float*)d_in[18];
    const float* gwhh  = (const float*)d_in[19];
    const float* gbih  = (const float*)d_in[20];
    const float* gbhh  = (const float*)d_in[21];
    const float* g1wl  = (const float*)d_in[22];
    const float* g1bl  = (const float*)d_in[23];
    const float* g1wr  = (const float*)d_in[24];
    const float* g1br  = (const float*)d_in[25];
    const float* g1att = (const float*)d_in[26];
    const float* g1bias= (const float*)d_in[27];
    const float* g2wl  = (const float*)d_in[28];
    const float* g2bl  = (const float*)d_in[29];
    const float* g2wr  = (const float*)d_in[30];
    const float* g2br  = (const float*)d_in[31];
    const float* g2att = (const float*)d_in[32];
    const float* g2bias= (const float*)d_in[33];
    float* out = (float*)d_out;   // [B, 39, 64] f32

    // ---- workspace bump allocator with region reuse -----------------------
    char* ws = (char*)d_ws;
    size_t off = 0;
    auto alloc = [&](size_t bytes) -> char* {
        char* p = ws + off;
        off = (off + bytes + 255) & ~(size_t)255;
        return p;
    };
    const size_t szGF  = (size_t)NTOT * K1P * 2;   // gFeat f16
    const size_t szX1  = (size_t)NTOT * F1  * 4;   // xl1 / xr1 / out1 f32
    const size_t szH1  = (size_t)NTOT * F1  * 2;   // h1 f16
    const size_t szX2  = (size_t)NTOT * F2  * 4;   // xl2 / xr2 f32
    const size_t szL1  = (size_t)ETOT * H1N * 4;
    const size_t szL2  = (size_t)ETOT * 4;
    const size_t szM1  = (size_t)NTOT * H1N * 4;
    const size_t szM2  = (size_t)NTOT * 4;

    _Float16* gFeat = (_Float16*)alloc(szGF);
    _Float16* w1lT  = (_Float16*)alloc((size_t)F1 * K1P * 2);
    _Float16* w1rT  = (_Float16*)alloc((size_t)F1 * K1P * 2);
    _Float16* w2lT  = (_Float16*)alloc((size_t)F2 * K2P * 2);
    _Float16* w2rT  = (_Float16*)alloc((size_t)F2 * K2P * 2);
    char* regC = alloc(szX1);                       // xl1, later xl2/xr2/logit2/m2/s2
    char* regD = alloc(szX1 > szH1 ? szX1 : szH1);  // xr1, later h1
    float*    logit1 = (float*)   alloc(szL1);
    unsigned* m1     = (unsigned*)alloc(szM1);
    float*    s1     = (float*)   alloc(szM1);
    float*    out1   = (float*)   alloc(szX1);

    float* xl1 = (float*)regC;
    float* xr1 = (float*)regD;
    // phase-2 aliases inside regC (dead xl1) and regD (dead xr1)
    float*    xl2    = (float*)regC;
    float*    xr2    = (float*)(regC + szX2);
    float*    logit2 = (float*)(regC + 2 * szX2);
    unsigned* m2     = (unsigned*)(regC + 2 * szX2 + ((szL2 + 255) & ~(size_t)255));
    float*    s2     = (float*)((char*)m2 + ((szM2 + 255) & ~(size_t)255));
    _Float16* h1     = (_Float16*)regD;

    // ---- stage 0: features ------------------------------------------------
    conv_gru_kernel<<<BATCH, 256, 0, stream>>>(
        ve, ac, man, mask, c1w, c1b, bn1g, bn1b, bn1m, bn1v,
        c2w, c2b, bn2g, bn2b, bn2m, bn2v, gwih, gwhh, gbih, gbhh, gFeat);
    man_feat_kernel<<<(unsigned)cdivz((size_t)NTOT * K1P, 256), 256, 0, stream>>>(
        man, mask, gFeat);
    convert_wT_kernel<<<(unsigned)cdivz((size_t)F1 * K1P, 256), 256, 0, stream>>>(
        g1wl, w1lT, 78, F1, K1P);
    convert_wT_kernel<<<(unsigned)cdivz((size_t)F1 * K1P, 256), 256, 0, stream>>>(
        g1wr, w1rT, 78, F1, K1P);
    convert_wT_kernel<<<(unsigned)cdivz((size_t)F2 * K2P, 256), 256, 0, stream>>>(
        g2wl, w2lT, K2P, F2, K2P);
    convert_wT_kernel<<<(unsigned)cdivz((size_t)F2 * K2P, 256), 256, 0, stream>>>(
        g2wr, w2rT, K2P, F2, K2P);

    // ---- stage 1: GATv2 layer 1 (78 -> 8x32) ------------------------------
    wmma_dual_gemm_kernel<<<dim3(NTOT / 16, F1 / 64), 32, 0, stream>>>(
        gFeat, w1lT, w1rT, g1bl, g1br, xl1, xr1, F1, K1P);
    fill_u32_kernel<<<(unsigned)cdivz((size_t)NTOT * H1N, 256), 256, 0, stream>>>(
        m1, 0u, (size_t)NTOT * H1N);
    fill_u32_kernel<<<(unsigned)cdivz((size_t)NTOT * H1N, 256), 256, 0, stream>>>(
        (unsigned*)s1, 0u, (size_t)NTOT * H1N);
    init_bias_kernel<<<(unsigned)cdivz((size_t)NTOT * F1, 256), 256, 0, stream>>>(
        out1, g1bias, NTOT, F1);
    gat_logit_max_kernel<<<(unsigned)cdivz((size_t)ETOT * H1N, 256), 256, 0, stream>>>(
        eidx, xl1, xr1, g1att, logit1, m1, H1N, D1);
    gat_expsum_kernel<<<(unsigned)cdivz((size_t)ETOT * H1N, 256), 256, 0, stream>>>(
        eidx, logit1, m1, s1, H1N);
    gat_aggregate_kernel<<<(unsigned)cdivz((size_t)ETOT * (F1 / 4), 256), 256, 0, stream>>>(
        eidx, logit1, s1, xl1, out1, H1N, D1);
    elu_f16_kernel<<<(unsigned)cdivz((size_t)NTOT * F1, 256), 256, 0, stream>>>(
        out1, h1, (size_t)NTOT * F1);

    // ---- stage 2: GATv2 layer 2 (256 -> 64), aggregate into d_out ---------
    wmma_dual_gemm_kernel<<<dim3(NTOT / 16, F2 / 64), 32, 0, stream>>>(
        h1, w2lT, w2rT, g2bl, g2br, xl2, xr2, F2, K2P);
    fill_u32_kernel<<<(unsigned)cdivz((size_t)NTOT, 256), 256, 0, stream>>>(
        m2, 0u, (size_t)NTOT);
    fill_u32_kernel<<<(unsigned)cdivz((size_t)NTOT, 256), 256, 0, stream>>>(
        (unsigned*)s2, 0u, (size_t)NTOT);
    init_bias_kernel<<<(unsigned)cdivz((size_t)NTOT * F2, 256), 256, 0, stream>>>(
        out, g2bias, NTOT, F2);
    gat_logit_max_kernel<<<(unsigned)cdivz((size_t)ETOT, 256), 256, 0, stream>>>(
        eidx, xl2, xr2, g2att, logit2, m2, 1, F2);
    gat_expsum_kernel<<<(unsigned)cdivz((size_t)ETOT, 256), 256, 0, stream>>>(
        eidx, logit2, m2, s2, 1);
    gat_aggregate_kernel<<<(unsigned)cdivz((size_t)ETOT * (F2 / 4), 256), 256, 0, stream>>>(
        eidx, logit2, s2, xl2, out, 1, F2);
}
